// SelfAttention_7430293422332
// MI455X (gfx1250) — matrix operand order repaired and checked by reading, not compiled, not run on hardware
//
#include <hip/hip_runtime.h>
#include <hip/hip_bf16.h>
#include <math.h>

typedef __bf16 bf16;
typedef __attribute__((ext_vector_type(16))) __bf16 v16bf;
typedef __attribute__((ext_vector_type(8)))  float  v8f;

#define B_   2
#define L_   2048
#define DIM_ 4096
#define H_   32
#define NK_  8
#define HD_  128
#define NREP_ (H_ / NK_)

union Frag16 { v16bf v; uint4 u[2]; };

// ---------------------------------------------------------------------------
// CDNA5 async global->LDS copy (16B per lane), tracked by ASYNCcnt.
// GV addressing: VDST = LDS byte address, VADDR = 64-bit global address.
// ---------------------------------------------------------------------------
__device__ __forceinline__ void async_cp_b128(void* lds_dst, const void* gsrc) {
  unsigned l = (unsigned)(unsigned long long)lds_dst;  // low 32b = LDS offset
  asm volatile("global_load_async_to_lds_b128 %0, %1, off"
               :: "v"(l), "v"(gsrc)
               : "memory");
}
__device__ __forceinline__ void wait_async0() {
  asm volatile("s_wait_asynccnt 0x0" ::: "memory");
}

// ---------------------------------------------------------------------------
// fp32 -> bf16 conversion (memory-bound; vectorized x4)
// ---------------------------------------------------------------------------
__global__ void cvt_f32_to_bf16(const float* __restrict__ src,
                                bf16* __restrict__ dst, int n) {
  int i = (blockIdx.x * blockDim.x + threadIdx.x) * 4;
  if (i < n) {
    float4 f = *(const float4*)(src + i);
    dst[i + 0] = (bf16)f.x;
    dst[i + 1] = (bf16)f.y;
    dst[i + 2] = (bf16)f.z;
    dst[i + 3] = (bf16)f.w;
  }
}

// ---------------------------------------------------------------------------
// RoPE applied in-place on a bf16 (B, L, heads, HD) buffer.
// angles: (L, HD/2) fp32, pairs are interleaved (even, odd) dims.
// ---------------------------------------------------------------------------
__global__ void rope_bf16(bf16* __restrict__ buf, const float* __restrict__ ang,
                          int heads, int total_pairs) {
  int t = blockIdx.x * blockDim.x + threadIdx.x;
  if (t >= total_pairs) return;
  int dp   = t & 63;          // HD/2 = 64
  int rest = t >> 6;
  int h = rest % heads; rest /= heads;
  int l = rest % L_;
  int b = rest / L_;
  float a = ang[l * 64 + dp];
  float c = __cosf(a), s = __sinf(a);
  size_t base = ((size_t)(b * L_ + l) * heads + h) * HD_ + dp * 2;
  float xr = (float)buf[base], xi = (float)buf[base + 1];
  buf[base]     = (bf16)(xr * c - xi * s);
  buf[base + 1] = (bf16)(xr * s + xi * c);
}

// ---------------------------------------------------------------------------
// bf16 WMMA GEMM: C[M,N] = A[M,K] @ B[K,N], A/B row-major bf16.
// Block tile 128x128x32, 256 threads = 8 waves (4 x 2), each wave 32x64.
// A staged via async global->LDS; B transposed to n-major through VGPRs.
// Output type selected at compile time (no runtime branch in epilogue).
// ---------------------------------------------------------------------------
#define BM 128
#define BN 128
#define BK 32
#define BKP 40   // padded LDS stride (bf16 elems) to spread banks

template <bool F32OUT>
__global__ __launch_bounds__(256)
void gemm_bf16_wmma(const bf16* __restrict__ A, const bf16* __restrict__ Bm,
                    void* __restrict__ Cptr, int M, int N, int K) {
  __shared__ __align__(16) bf16 sA[BM * BKP];   // (m, k) row-major, padded
  __shared__ __align__(16) bf16 sB[BN * BKP];   // (n, k) n-major (transposed), padded

  const int tid  = threadIdx.x;
  const int lane = tid & 31;
  const int wv   = tid >> 5;
  const int wM   = wv & 3;    // 4 row-slices of 32
  const int wN   = wv >> 2;   // 2 col-slices of 64
  const int hf   = lane >> 4; // lane half (wave32 WMMA layout)
  const int l16  = lane & 15;
  const int bMr  = blockIdx.y * BM;
  const int bNc  = blockIdx.x * BN;

  v8f acc[2][4];
#pragma unroll
  for (int i = 0; i < 2; ++i)
#pragma unroll
    for (int j = 0; j < 4; ++j)
#pragma unroll
      for (int e = 0; e < 8; ++e) acc[i][j][e] = 0.0f;

  for (int k0 = 0; k0 < K; k0 += BK) {
    // Stage A asynchronously: 128x32 bf16 = 512 chunks of 16B, 2 per thread.
#pragma unroll
    for (int i = 0; i < 2; ++i) {
      int c = tid + i * 256;
      int row = c >> 2, kc = (c & 3) * 8;
      async_cp_b128(&sA[row * BKP + kc],
                    &A[(size_t)(bMr + row) * K + k0 + kc]);
    }
    // Stage B transposed to n-major: read 16B rows, scatter 8 bf16.
#pragma unroll
    for (int i = 0; i < 2; ++i) {
      int c = tid + i * 256;
      int k = c >> 4, nc = (c & 15) * 8;
      uint4 t = *(const uint4*)&Bm[(size_t)(k0 + k) * N + bNc + nc];
      const bf16* tv = (const bf16*)&t;
#pragma unroll
      for (int j = 0; j < 8; ++j) sB[(nc + j) * BKP + k] = tv[j];
    }
    wait_async0();
    __syncthreads();

    Frag16 af[2], bf[4];
#pragma unroll
    for (int tm = 0; tm < 2; ++tm) {
      int row = wM * 32 + tm * 16 + l16;
      af[tm].u[0] = *(const uint4*)&sA[row * BKP + 16 * hf];        // K = hf*16 .. +7
      af[tm].u[1] = *(const uint4*)&sA[row * BKP + 16 * hf + 8];   // K = hf*16+8 .. +15
    }
#pragma unroll
    for (int tn = 0; tn < 4; ++tn) {
      int n = wN * 64 + tn * 16 + l16;
      bf[tn].u[0] = *(const uint4*)&sB[n * BKP + 16 * hf];         // K = hf*16 .. +15
      bf[tn].u[1] = *(const uint4*)&sB[n * BKP + 16 * hf + 8];
    }
#pragma unroll
    for (int tm = 0; tm < 2; ++tm)
#pragma unroll
      for (int tn = 0; tn < 4; ++tn)
        acc[tm][tn] = __builtin_amdgcn_wmma_f32_16x16x32_bf16(
            false, af[tm].v, false, bf[tn].v, (short)0, acc[tm][tn],
            false, false);
    __syncthreads();
  }

  // Epilogue: D layout = VGPR r -> row (r + 8*hf), lane%16 -> col.
#pragma unroll
  for (int tm = 0; tm < 2; ++tm) {
#pragma unroll
    for (int tn = 0; tn < 4; ++tn) {
      int col = bNc + wN * 64 + tn * 16 + l16;
#pragma unroll
      for (int r = 0; r < 8; ++r) {
        int row = bMr + wM * 32 + tm * 16 + r + 8 * hf;
        if constexpr (F32OUT) {
          ((float*)Cptr)[(size_t)row * N + col] = acc[tm][tn][r];
        } else {
          ((bf16*)Cptr)[(size_t)row * N + col] = (bf16)acc[tm][tn][r];
        }
      }
    }
  }
}

// ---------------------------------------------------------------------------
// Causal flash attention with GQA.  Grid: (L/64, B*H). 128 threads = 4 waves,
// each wave owns 16 query rows. KV processed in blocks of 32, WMMA for both
// QK^T and PV, online softmax in fp32. K staged via async global->LDS.
// ---------------------------------------------------------------------------
__global__ __launch_bounds__(128)
void flash_attn_wmma(const bf16* __restrict__ Q, const bf16* __restrict__ Kb,
                     const bf16* __restrict__ Vb, bf16* __restrict__ O) {
  __shared__ __align__(16) bf16 sK[32 * HD_];        // (kv, d) row-major: B-op for scores
  __shared__ __align__(16) bf16 sVT[HD_ * 32];       // (d, kv): B-op for PV
  __shared__ __align__(16) bf16 sP[4][16 * 32];      // per-wave P tile (C->A relayout)

  const int tid  = threadIdx.x;
  const int lane = tid & 31;
  const int wv   = tid >> 5;
  const int hf   = lane >> 4;
  const int l16  = lane & 15;
  const int qblk = blockIdx.x;
  const int bh   = blockIdx.y;
  const int b    = bh / H_;
  const int h    = bh % H_;
  const int kvh  = h / NREP_;
  const int qbase = qblk * 64 + wv * 16;
  const float scale = 0.08838834764831845f; // 1/sqrt(128)

  // Q fragments: 16 rows x HD, split into 4 K-chunks of 32 (A-layout).
  Frag16 qf[4];
  {
    const bf16* qp = Q + ((size_t)(b * L_ + qbase + l16) * H_ + h) * HD_;
#pragma unroll
    for (int ck = 0; ck < 4; ++ck) {
      qf[ck].u[0] = *(const uint4*)&qp[ck * 32 + 16 * hf];
      qf[ck].u[1] = *(const uint4*)&qp[ck * 32 + 16 * hf + 8];
    }
  }

  v8f acc[8];
#pragma unroll
  for (int i = 0; i < 8; ++i)
#pragma unroll
    for (int e = 0; e < 8; ++e) acc[i][e] = 0.0f;
  float m[8], ls[8];
#pragma unroll
  for (int r = 0; r < 8; ++r) { m[r] = -INFINITY; ls[r] = 0.0f; }

  const int nkv = qblk * 2 + 2;   // causal: kv cols up to qblk*64+63
  for (int kb = 0; kb < nkv; ++kb) {
    const int kvbase = kb * 32;
    // Stage K (async, row-major) and V (transposed through VGPRs).
#pragma unroll
    for (int i = 0; i < 4; ++i) {
      int c  = tid + i * 128;
      int kv = c >> 4, dc = (c & 15) * 8;
      size_t g = ((size_t)(b * L_ + kvbase + kv) * NK_ + kvh) * HD_ + dc;
      async_cp_b128(&sK[kv * HD_ + dc], &Kb[g]);
      uint4 t = *(const uint4*)&Vb[g];
      const bf16* tv = (const bf16*)&t;
#pragma unroll
      for (int j = 0; j < 8; ++j) sVT[(dc + j) * 32 + kv] = tv[j];
    }
    wait_async0();
    __syncthreads();

    // Scores: S(16x32) = Q(16x128) @ K^T. Two 16-col tiles, 4 K-chunks each.
    v8f s0, s1;
#pragma unroll
    for (int e = 0; e < 8; ++e) { s0[e] = 0.0f; s1[e] = 0.0f; }
#pragma unroll
    for (int ck = 0; ck < 4; ++ck) {
      Frag16 bk0, bk1;
      bk0.u[0] = *(const uint4*)&sK[l16 * HD_ + ck * 32 + 16 * hf];
      bk0.u[1] = *(const uint4*)&sK[l16 * HD_ + ck * 32 + 16 * hf + 8];
      bk1.u[0] = *(const uint4*)&sK[(16 + l16) * HD_ + ck * 32 + 16 * hf];
      bk1.u[1] = *(const uint4*)&sK[(16 + l16) * HD_ + ck * 32 + 16 * hf + 8];
      s0 = __builtin_amdgcn_wmma_f32_16x16x32_bf16(false, qf[ck].v, false,
                                                   bk0.v, (short)0, s0, false, false);
      s1 = __builtin_amdgcn_wmma_f32_16x16x32_bf16(false, qf[ck].v, false,
                                                   bk1.v, (short)0, s1, false, false);
    }

    // Online softmax (rows live in VGPR index, cols across 16-lane halves).
#pragma unroll
    for (int r = 0; r < 8; ++r) {
      int row = qbase + r + 8 * hf;
      float v0 = (kvbase + l16      <= row) ? s0[r] * scale : -INFINITY;
      float v1 = (kvbase + 16 + l16 <= row) ? s1[r] * scale : -INFINITY;
      float mx = fmaxf(v0, v1);
#pragma unroll
      for (int off = 1; off < 16; off <<= 1)
        mx = fmaxf(mx, __shfl_xor(mx, off, 32));
      float mn = fmaxf(m[r], mx);
      float p0 = __expf(v0 - mn);
      float p1 = __expf(v1 - mn);
      float sum = p0 + p1;
#pragma unroll
      for (int off = 1; off < 16; off <<= 1)
        sum += __shfl_xor(sum, off, 32);
      float alpha = __expf(m[r] - mn);
      m[r]  = mn;
      ls[r] = ls[r] * alpha + sum;
#pragma unroll
      for (int dt = 0; dt < 8; ++dt) acc[dt][r] *= alpha;
      // Write P in row-major to per-wave LDS for C->A relayout.
      sP[wv][(r + 8 * hf) * 32 + l16]      = (bf16)p0;
      sP[wv][(r + 8 * hf) * 32 + 16 + l16] = (bf16)p1;
    }
    asm volatile("s_wait_dscnt 0x0" ::: "memory");   // wave-local LDS RAW

    // P as A-fragment (16x32), V^T tiles as B: O += P @ V.
    Frag16 pf;
    pf.u[0] = *(const uint4*)&sP[wv][l16 * 32 + 16 * hf];
    pf.u[1] = *(const uint4*)&sP[wv][l16 * 32 + 16 * hf + 8];
#pragma unroll
    for (int dt = 0; dt < 8; ++dt) {
      Frag16 bv;
      bv.u[0] = *(const uint4*)&sVT[(dt * 16 + l16) * 32 + 16 * hf];
      bv.u[1] = *(const uint4*)&sVT[(dt * 16 + l16) * 32 + 16 * hf + 8];
      acc[dt] = __builtin_amdgcn_wmma_f32_16x16x32_bf16(
          false, pf.v, false, bv.v, (short)0, acc[dt], false, false);
    }
    __syncthreads();
  }

  // Normalize and store (b, l, h, d) bf16.
#pragma unroll
  for (int dt = 0; dt < 8; ++dt) {
    int col = dt * 16 + l16;
#pragma unroll
    for (int r = 0; r < 8; ++r) {
      int row = qbase + r + 8 * hf;
      O[((size_t)(b * L_ + row) * H_ + h) * HD_ + col] = (bf16)(acc[dt][r] / ls[r]);
    }
  }
}

// ---------------------------------------------------------------------------
extern "C" void kernel_launch(void* const* d_in, const int* in_sizes, int n_in,
                              void* d_out, int out_size, void* d_ws, size_t ws_size,
                              hipStream_t stream) {
  (void)in_sizes; (void)n_in; (void)out_size; (void)ws_size;
  const float* x   = (const float*)d_in[0];
  const float* ang = (const float*)d_in[1];
  const float* Wq  = (const float*)d_in[2];
  const float* Wk  = (const float*)d_in[3];
  const float* Wv  = (const float*)d_in[4];
  const float* Wo  = (const float*)d_in[5];
  float* out = (float*)d_out;   // start_pos = 0; caches unused (not returned)

  char* ws = (char*)d_ws;
  size_t off = 0;
  auto alloc = [&](size_t bytes) {
    void* p = ws + off;
    off += (bytes + 255) & ~(size_t)255;
    return p;
  };
  const size_t M = (size_t)B_ * L_;           // 4096
  bf16* xbf = (bf16*)alloc(M * DIM_ * 2);
  bf16* wqb = (bf16*)alloc((size_t)DIM_ * H_ * HD_ * 2);
  bf16* wkb = (bf16*)alloc((size_t)DIM_ * NK_ * HD_ * 2);
  bf16* wvb = (bf16*)alloc((size_t)DIM_ * NK_ * HD_ * 2);
  bf16* wob = (bf16*)alloc((size_t)H_ * HD_ * DIM_ * 2);
  bf16* qb  = (bf16*)alloc(M * H_ * HD_ * 2);
  bf16* kb  = (bf16*)alloc(M * NK_ * HD_ * 2);
  bf16* vb  = (bf16*)alloc(M * NK_ * HD_ * 2);
  bf16* ab  = (bf16*)alloc(M * H_ * HD_ * 2);

  auto cvt = [&](const float* s, bf16* d, size_t n) {
    int blocks = (int)((n / 4 + 255) / 256);
    cvt_f32_to_bf16<<<dim3(blocks), dim3(256), 0, stream>>>(s, d, (int)n);
  };
  cvt(x,  xbf, M * DIM_);
  cvt(Wq, wqb, (size_t)DIM_ * H_ * HD_);
  cvt(Wk, wkb, (size_t)DIM_ * NK_ * HD_);
  cvt(Wv, wvb, (size_t)DIM_ * NK_ * HD_);
  cvt(Wo, wob, (size_t)H_ * HD_ * DIM_);

  dim3 blk(256);
  // Q/K/V projections (bf16 out)
  gemm_bf16_wmma<false><<<dim3((H_ * HD_) / BN, (int)M / BM), blk, 0, stream>>>(
      xbf, wqb, (void*)qb, (int)M, H_ * HD_, DIM_);
  gemm_bf16_wmma<false><<<dim3((NK_ * HD_) / BN, (int)M / BM), blk, 0, stream>>>(
      xbf, wkb, (void*)kb, (int)M, NK_ * HD_, DIM_);
  gemm_bf16_wmma<false><<<dim3((NK_ * HD_) / BN, (int)M / BM), blk, 0, stream>>>(
      xbf, wvb, (void*)vb, (int)M, NK_ * HD_, DIM_);

  // RoPE on Q and K
  {
    int pq = B_ * L_ * H_ * (HD_ / 2);
    rope_bf16<<<dim3((pq + 255) / 256), dim3(256), 0, stream>>>(qb, ang, H_, pq);
    int pk = B_ * L_ * NK_ * (HD_ / 2);
    rope_bf16<<<dim3((pk + 255) / 256), dim3(256), 0, stream>>>(kb, ang, NK_, pk);
  }

  // Flash attention
  flash_attn_wmma<<<dim3(L_ / 64, B_ * H_), dim3(128), 0, stream>>>(qb, kb, vb, ab);

  // Output projection -> fp32
  gemm_bf16_wmma<true><<<dim3(DIM_ / BN, (int)M / BM), blk, 0, stream>>>(
      ab, wob, (void*)out, (int)M, DIM_, H_ * HD_);
}
